// VAECOV1_75359496176088
// MI455X (gfx1250) — matrix operand order repaired and checked by reading, not compiled, not run on hardware
//
#include <hip/hip_runtime.h>

// ---------------------------------------------------------------------------
// CDNA5 (gfx1250) VAE-with-covariance forward.
// GEMMs via v_wmma_f32_16x16x32_bf16; tile feeds via GLOBAL_LOAD_ASYNC_TO_LDS
// (ASYNCcnt DMA) when the builtin exists, else register-staged b128 copies.
// ---------------------------------------------------------------------------

typedef __attribute__((ext_vector_type(16))) __bf16 v16bf;
typedef __attribute__((ext_vector_type(8)))  float  v8f;
typedef int v4i __attribute__((__vector_size__(16)));  // matches builtin proto

struct Frag { union { v16bf v; uint4 q[2]; }; };

#if defined(__has_builtin)
#  if __has_builtin(__builtin_amdgcn_global_load_async_to_lds_b128)
#    define USE_ASYNC_LDS 1
#  endif
#endif
#ifndef USE_ASYNC_LDS
#  define USE_ASYNC_LDS 0
#endif

#define GLOBAL_AS __attribute__((address_space(1)))
#define LDS_AS    __attribute__((address_space(3)))

__device__ __forceinline__ GLOBAL_AS v4i* gcast(const void* p) {
  return (GLOBAL_AS v4i*)(unsigned long long)p;
}
// Flat LDS address: addr[31:0] is the LDS byte offset (aperture mapping).
__device__ __forceinline__ LDS_AS v4i* lcast(void* p) {
  return (LDS_AS v4i*)(unsigned int)(unsigned long long)p;
}

__device__ __forceinline__ unsigned short f2bf(float f) {
  unsigned int u = __float_as_uint(f);
  u += 0x7FFFu + ((u >> 16) & 1u);            // round-to-nearest-even
  return (unsigned short)(u >> 16);
}

// ---------------------------------------------------------------------------
// f32 -> bf16, row-major, zero-padded (for the activation/A side).
// ---------------------------------------------------------------------------
__global__ void convert_pad_bf16(const float* __restrict__ in,
                                 unsigned short* __restrict__ out,
                                 int in_rows, int in_cols,
                                 int out_rows, int out_cols) {
  int idx = blockIdx.x * blockDim.x + threadIdx.x;
  if (idx >= out_rows * out_cols) return;
  int r = idx / out_cols, c = idx - r * out_cols;
  float v = (r < in_rows && c < in_cols) ? in[(size_t)r * in_cols + c] : 0.0f;
  out[idx] = f2bf(v);
}

// ---------------------------------------------------------------------------
// f32 (K_in x N_in, row-major) -> bf16 TRANSPOSED (N_out x K_out), zero-padded.
// Makes the GEMM B-tile a straight row copy (same layout as the A tile).
// ---------------------------------------------------------------------------
__global__ void convert_pad_bf16_T(const float* __restrict__ in,
                                   unsigned short* __restrict__ out,
                                   int K_in, int N_in, int K_out, int N_out) {
  int idx = blockIdx.x * blockDim.x + threadIdx.x;
  if (idx >= N_out * K_out) return;
  int n = idx / K_out, k = idx - n * K_out;
  float v = (k < K_in && n < N_in) ? in[(size_t)k * N_in + n] : 0.0f;
  out[idx] = f2bf(v);
}

// ---------------------------------------------------------------------------
// Tiled bf16 WMMA GEMM: Out(MxN) = epilogue(A(MxK) @ Bt(NpadxK)^T + bias)
//   EPI = 0 : f32 store   EPI = 1 : relu -> bf16   EPI = 2 : sigmoid -> f32
// Block tile 128x128, K-step 32, 8 waves (2Mx4N), wave tile 64x32 (8 WMMA).
// Double-buffered LDS, both tiles identical padded layout (stride 40 shorts
// = 20 banks: conflict-free b128 fragment reads per 16-lane group).
// Requires: M % 128 == 0, K % 32 == 0, Bt has ceil(N/128)*128 rows.
// ---------------------------------------------------------------------------
template <int EPI>
__global__ __launch_bounds__(256)
void gemm_bf16_wmma(const unsigned short* __restrict__ A,
                    const unsigned short* __restrict__ Bt,
                    const float* __restrict__ bias,
                    void* __restrict__ OutP,
                    int M, int N, int K) {
  constexpr int BM = 128, BN = 128, BK = 32, STR = 40;
  __shared__ __align__(16) unsigned short As[2][BM * STR];
  __shared__ __align__(16) unsigned short Bs[2][BN * STR];

  const int tid  = threadIdx.x;
  const int wave = tid >> 5, lane = tid & 31;
  const int wm = wave & 1, wn = wave >> 1;      // 2 (M) x 4 (N) wave grid
  const int half = lane >> 4, l16 = lane & 15;
  const int m0 = blockIdx.y * BM, n0 = blockIdx.x * BN;

  // 128 rows x 32 bf16 per tile -> 512 uint4 chunks; 2 per thread per tile
  const int row0 = tid >> 2,         off0 = (tid & 3) * 8;
  const int row1 = (tid + 256) >> 2, off1 = ((tid + 256) & 3) * 8;

  const int nk = K / BK;

#if USE_ASYNC_LDS
  auto copy_tile = [&](int kt, int s) {
    __builtin_amdgcn_global_load_async_to_lds_b128(
        gcast(A + (size_t)(m0 + row0) * K + kt * BK + off0),
        lcast(&As[s][row0 * STR + off0]), 0, 0);
    __builtin_amdgcn_global_load_async_to_lds_b128(
        gcast(A + (size_t)(m0 + row1) * K + kt * BK + off1),
        lcast(&As[s][row1 * STR + off1]), 0, 0);
    __builtin_amdgcn_global_load_async_to_lds_b128(
        gcast(Bt + (size_t)(n0 + row0) * K + kt * BK + off0),
        lcast(&Bs[s][row0 * STR + off0]), 0, 0);
    __builtin_amdgcn_global_load_async_to_lds_b128(
        gcast(Bt + (size_t)(n0 + row1) * K + kt * BK + off1),
        lcast(&Bs[s][row1 * STR + off1]), 0, 0);
  };
#else
  uint4 ra0, ra1, rb0, rb1;
  auto gload = [&](int kt) {
    ra0 = *reinterpret_cast<const uint4*>(A + (size_t)(m0 + row0) * K + kt * BK + off0);
    ra1 = *reinterpret_cast<const uint4*>(A + (size_t)(m0 + row1) * K + kt * BK + off1);
    rb0 = *reinterpret_cast<const uint4*>(Bt + (size_t)(n0 + row0) * K + kt * BK + off0);
    rb1 = *reinterpret_cast<const uint4*>(Bt + (size_t)(n0 + row1) * K + kt * BK + off1);
  };
  auto lstore = [&](int s) {
    *reinterpret_cast<uint4*>(&As[s][row0 * STR + off0]) = ra0;
    *reinterpret_cast<uint4*>(&As[s][row1 * STR + off1]) = ra1;
    *reinterpret_cast<uint4*>(&Bs[s][row0 * STR + off0]) = rb0;
    *reinterpret_cast<uint4*>(&Bs[s][row1 * STR + off1]) = rb1;
  };
#endif

  v8f zero;
#pragma unroll
  for (int e = 0; e < 8; ++e) zero[e] = 0.0f;
  v8f acc[4][2];
#pragma unroll
  for (int i = 0; i < 4; ++i)
#pragma unroll
    for (int j = 0; j < 2; ++j) acc[i][j] = zero;

#if USE_ASYNC_LDS
  copy_tile(0, 0);
  asm volatile("s_wait_asynccnt 0x0" ::: "memory");
  __syncthreads();
#else
  gload(0);
  lstore(0);
  __syncthreads();
#endif

  for (int kt = 0; kt < nk; ++kt) {
    const int s = kt & 1;
#if USE_ASYNC_LDS
    if (kt + 1 < nk) copy_tile(kt + 1, s ^ 1);   // DMA overlaps the WMMAs below
#else
    if (kt + 1 < nk) gload(kt + 1);
#endif
    if (kt + 2 < nk) {  // global_prefetch_b8 into L2 for the tile after next
      __builtin_prefetch(A + (size_t)(m0 + row0) * K + (kt + 2) * BK + off0, 0, 1);
      __builtin_prefetch(Bt + (size_t)(n0 + row0) * K + (kt + 2) * BK + off0, 0, 1);
    }

    Frag aF[4], bF[2];
#pragma unroll
    for (int tm = 0; tm < 4; ++tm) {
      const int m = wm * 64 + tm * 16 + l16;
      aF[tm].q[0] = *reinterpret_cast<const uint4*>(&As[s][m * STR + half * 8]);
      aF[tm].q[1] = *reinterpret_cast<const uint4*>(&As[s][m * STR + 16 + half * 8]);
    }
#pragma unroll
    for (int tn = 0; tn < 2; ++tn) {
      const int n = wn * 32 + tn * 16 + l16;
      bF[tn].q[0] = *reinterpret_cast<const uint4*>(&Bs[s][n * STR + half * 8]);
      bF[tn].q[1] = *reinterpret_cast<const uint4*>(&Bs[s][n * STR + 16 + half * 8]);
    }
#pragma unroll
    for (int tm = 0; tm < 4; ++tm)
#pragma unroll
      for (int tn = 0; tn < 2; ++tn)
        acc[tm][tn] = __builtin_amdgcn_wmma_f32_16x16x32_bf16(
            false, aF[tm].v, false, bF[tn].v, (short)0, acc[tm][tn], false, false);

#if USE_ASYNC_LDS
    asm volatile("s_wait_asynccnt 0x0" ::: "memory");
    __syncthreads();
#else
    if (kt + 1 < nk) lstore((kt + 1) & 1);
    __syncthreads();
#endif
  }

  // Epilogue: C VGPR i -> row = tile_row0 + half*8 + i, col = tile_col0 + l16
#pragma unroll
  for (int tm = 0; tm < 4; ++tm) {
    const int rbase = m0 + wm * 64 + tm * 16 + half * 8;
#pragma unroll
    for (int tn = 0; tn < 2; ++tn) {
      const int col = n0 + wn * 32 + tn * 16 + l16;
      if (col >= N) continue;
      const float bv = bias[col];
#pragma unroll
      for (int i = 0; i < 8; ++i) {
        float v = acc[tm][tn][i] + bv;
        const size_t oidx = (size_t)(rbase + i) * N + col;
        if (EPI == 1) {
          ((unsigned short*)OutP)[oidx] = f2bf(v > 0.0f ? v : 0.0f);
        } else if (EPI == 2) {
          ((float*)OutP)[oidx] = 1.0f / (1.0f + __expf(-v));
        } else {
          ((float*)OutP)[oidx] = v;
        }
      }
    }
  }
}

// ---------------------------------------------------------------------------
// Sampling: z = mu + A@eps, logvar = log(diag(A A^T)); the scattered A factor
// collapses to 8-element dots. Also emits zero-padded bf16 z (K=32).
// ---------------------------------------------------------------------------
__global__ void sample_kernel(const float* __restrict__ std1,  // B x 1024
                              const float* __restrict__ mu,    // B x 128
                              const float* __restrict__ eps,   // B x 128
                              float* __restrict__ zf,          // (B*8) x 16
                              float* __restrict__ logvar,      // (B*8) x 16
                              unsigned short* __restrict__ zb) // (B*8) x 32 bf16
{
  int idx = blockIdx.x * blockDim.x + threadIdx.x;  // over B*128
  if (idx >= 4096 * 128) return;
  int b = idx >> 7, r = idx & 127;
  int d = r & 15, s1 = r >> 4;
  const float* vp = std1 + (size_t)b * 1024 + r * 8;
  const float* ep = eps + (size_t)b * 128 + d;
  float sum = 0.0f, sq = 0.0f;
#pragma unroll
  for (int k = 0; k < 8; ++k) {
    float v = vp[k];
    sum += v * ep[k * 16];  // col = d + k*16
    sq  += v * v;
  }
  float zv = mu[(size_t)b * 128 + r] + sum;
  int ro = (b * 8 + s1) * 16 + d;
  zf[ro] = zv;
  logvar[ro] = __logf(sq);
  int zo = (b * 8 + s1) * 32 + d;
  zb[zo] = f2bf(zv);
  zb[zo + 16] = 0;  // K-pad 16..31
}

// ---------------------------------------------------------------------------
// Launcher
// ---------------------------------------------------------------------------
extern "C" void kernel_launch(void* const* d_in, const int* in_sizes, int n_in,
                              void* d_out, int out_size, void* d_ws, size_t ws_size,
                              hipStream_t stream) {
  (void)in_sizes; (void)n_in; (void)out_size; (void)ws_size;
  const float* x    = (const float*)d_in[0];
  const float* eps  = (const float*)d_in[1];
  const float* W1   = (const float*)d_in[2];
  const float* b1   = (const float*)d_in[3];
  const float* Wmu  = (const float*)d_in[4];
  const float* bmu  = (const float*)d_in[5];
  const float* Wstd = (const float*)d_in[6];
  const float* bstd = (const float*)d_in[7];
  const float* Wd1  = (const float*)d_in[8];
  const float* bd1  = (const float*)d_in[9];
  const float* Wd2  = (const float*)d_in[10];
  const float* bd2  = (const float*)d_in[11];

  constexpr int Bn = 4096, X = 784, XP = 800, XNP = 896;  // XNP: N-pad of 784
  constexpr int H = 512, ZS = 128, SS = 1024;             // Z*S, Z*S*S
  constexpr int BS = Bn * 8;                              // 32768
  constexpr int KZ = 32;                                  // padded latent K

  // d_out layout: recon | mu | logvar | z (float32, flat, return order)
  float* recon  = (float*)d_out;
  float* mu     = recon + (size_t)BS * X;
  float* logvar = mu + (size_t)Bn * ZS;
  float* zf     = logvar + (size_t)BS * 16;

  // workspace layout (256B aligned slots)
  unsigned char* ws = (unsigned char*)d_ws;
  size_t off = 0;
  auto take = [&](size_t bytes) -> unsigned char* {
    unsigned char* p = ws + off;
    off = (off + bytes + 255) & ~(size_t)255;
    return p;
  };
  unsigned short* xb    = (unsigned short*)take((size_t)Bn * XP * 2);   // A: B x 800
  unsigned short* W1T   = (unsigned short*)take((size_t)H * XP * 2);    // Bt: 512 x 800
  unsigned short* WmuT  = (unsigned short*)take((size_t)ZS * H * 2);    // Bt: 128 x 512
  unsigned short* WstdT = (unsigned short*)take((size_t)SS * H * 2);    // Bt: 1024 x 512
  unsigned short* Wd1T  = (unsigned short*)take((size_t)H * KZ * 2);    // Bt: 512 x 32
  unsigned short* Wd2T  = (unsigned short*)take((size_t)XNP * H * 2);   // Bt: 896 x 512
  unsigned short* hb    = (unsigned short*)take((size_t)Bn * H * 2);    // A: B x 512
  float*          std1  = (float*)take((size_t)Bn * SS * 4);
  unsigned short* zb    = (unsigned short*)take((size_t)BS * KZ * 2);   // A: BS x 32
  unsigned short* hdb   = (unsigned short*)take((size_t)BS * H * 2);    // A: BS x 512

  auto cgrid = [](size_t n) { return (unsigned)((n + 255) / 256); };

  // 1) bf16 conversions: activations row-major K-padded; weights TRANSPOSED
  //    (N x K) with both K padded to mult-of-32 and N padded to mult-of-128.
  convert_pad_bf16  <<<cgrid((size_t)Bn * XP),  256, 0, stream>>>(x,   xb,   Bn, X, Bn, XP);
  convert_pad_bf16_T<<<cgrid((size_t)H * XP),   256, 0, stream>>>(W1,   W1T,   X, H,  XP, H);
  convert_pad_bf16_T<<<cgrid((size_t)ZS * H),   256, 0, stream>>>(Wmu,  WmuT,  H, ZS, H,  ZS);
  convert_pad_bf16_T<<<cgrid((size_t)SS * H),   256, 0, stream>>>(Wstd, WstdT, H, SS, H,  SS);
  convert_pad_bf16_T<<<cgrid((size_t)H * KZ),   256, 0, stream>>>(Wd1,  Wd1T,  16, H, KZ, H);
  convert_pad_bf16_T<<<cgrid((size_t)XNP * H),  256, 0, stream>>>(Wd2,  Wd2T,  H, X,  H,  XNP);

  // 2) encoder: h = relu(x @ W1 + b1) -> bf16
  gemm_bf16_wmma<1><<<dim3(H / 128, Bn / 128), 256, 0, stream>>>(xb, W1T, b1, hb, Bn, H, XP);
  // 3) mu = h @ Wmu + bmu -> f32 (straight into d_out)
  gemm_bf16_wmma<0><<<dim3(1, Bn / 128), 256, 0, stream>>>(hb, WmuT, bmu, mu, Bn, ZS, H);
  // 4) std1 = h @ Wstd + bstd -> f32 workspace
  gemm_bf16_wmma<0><<<dim3(SS / 128, Bn / 128), 256, 0, stream>>>(hb, WstdT, bstd, std1, Bn, SS, H);
  // 5) sample: z, logvar, padded bf16 z
  sample_kernel<<<cgrid((size_t)Bn * ZS), 256, 0, stream>>>(std1, mu, eps, zf, logvar, zb);
  // 6) decoder: hd = relu(z @ Wd1 + bd1) -> bf16
  gemm_bf16_wmma<1><<<dim3(H / 128, BS / 128), 256, 0, stream>>>(zb, Wd1T, bd1, hdb, BS, H, KZ);
  // 7) recon = sigmoid(hd @ Wd2 + bd2) -> f32 (N=784 store-guarded)
  gemm_bf16_wmma<2><<<dim3(XNP / 128, BS / 128), 256, 0, stream>>>(hdb, Wd2T, bd2, recon, BS, X, H);
}
